// FocusedLinearAttention_33380485825294
// MI455X (gfx1250) — matrix operand order repaired and checked by reading, not compiled, not run on hardware
//
#include <hip/hip_runtime.h>
#include <hip/hip_bf16.h>

// ---------------------------------------------------------------------------
// FocusedLinearAttention for MI455X (gfx1250, wave32, WMMA bf16 16x16x32)
// B=32, C=512, H=W=32 (N=1024), heads=8, hd=64
// Round 4: ping-pong (unroll-by-2) register double buffering -> no fragment
//          register copies in the steady state (kills the v_dual_mov +
//          WMMA->VALU hazard v_nop bursts seen in round 3 asm).
// ---------------------------------------------------------------------------

typedef __attribute__((ext_vector_type(16))) __bf16 bf16x16;
typedef __attribute__((ext_vector_type(8)))  __bf16 bf16x8;
typedef __attribute__((ext_vector_type(8)))  float  f32x8;

#define B_    32
#define C_    512
#define N_    1024
#define HEADS 8
#define HD    64
#define C3    1536

// ---------------- WMMA fragment helpers (ISA 7.12.2 layouts) ----------------
// A (16x32 bf16): lane L: row m = L&15, h = L>>4.
//   VGPR0..3 hold K = h*8 .. h*8+7 ; VGPR4..7 hold K = 16+h*8 .. 16+h*8+7
__device__ __forceinline__ bf16x16 load_a_frag(const __bf16* __restrict__ row,
                                               int k0, int h) {
  bf16x8 lo = *(const bf16x8*)(row + k0 + h * 8);
  bf16x8 hi = *(const bf16x8*)(row + k0 + 16 + h * 8);
  return __builtin_shufflevector(lo, hi, 0,1,2,3,4,5,6,7,8,9,10,11,12,13,14,15);
}
// B (32x16 bf16): lane L: col n = L&15, h = L>>4; K = h*8 .. h*8+7, 16+h*8 .. 16+h*8+7
// (operand stored with K innermost, i.e. as B^T row-major)
__device__ __forceinline__ bf16x16 load_b_frag(const __bf16* __restrict__ col,
                                               int k0, int h) {
  bf16x8 lo = *(const bf16x8*)(col + k0 + h * 8);
  bf16x8 hi = *(const bf16x8*)(col + k0 + 16 + h * 8);
  return __builtin_shufflevector(lo, hi, 0,1,2,3,4,5,6,7,8,9,10,11,12,13,14,15);
}
__device__ __forceinline__ f32x8 wmma_bf16(bf16x16 a, bf16x16 b, f32x8 c) {
  return __builtin_amdgcn_wmma_f32_16x16x32_bf16(false, a, false, b,
                                                 (short)0, c, false, false);
}

// ---------------- small converters -----------------------------------------
__global__ void w_to_bf16_kernel(const float* __restrict__ w,
                                 __bf16* __restrict__ wb, int n) {
  int t = blockIdx.x * 256 + threadIdx.x;
  if (t < n) wb[t] = (__bf16)w[t];
}

// x (B,C,N) f32 -> xb (B,N,C) bf16, 32x32 LDS-tiled transpose
__global__ void x_transpose_kernel(const float* __restrict__ x,
                                   __bf16* __restrict__ xb) {
  __shared__ float tile[32][33];
  int blk  = blockIdx.x;                 // B * (C/32) * (N/32) = 32*16*32
  int b    = blk >> 9;
  int rest = blk & 511;
  int c0   = (rest >> 5) * 32;
  int n0   = (rest & 31) * 32;
  int tx = threadIdx.x & 31;
  int ty = threadIdx.x >> 5;             // 8 rows / pass
#pragma unroll
  for (int p = 0; p < 4; ++p) {
    int i = ty + p * 8;
    tile[i][tx] = x[((size_t)(b * C_ + c0 + i)) * N_ + n0 + tx];
  }
  __syncthreads();
#pragma unroll
  for (int p = 0; p < 4; ++p) {
    int i = ty + p * 8;                  // n within tile
    xb[((size_t)(b * N_ + n0 + i)) * C_ + c0 + tx] = (__bf16)tile[tx][i];
  }
}

// ---------------- qkv GEMM: qkv[b,n,o] = sum_c W[o,c] X[b,n,c] --------------
// wave tile: 32 (o) x 64 (n), K = 512, ping-pong register pipeline
__global__ void __launch_bounds__(256, 1)
qkv_gemm_kernel(const __bf16* __restrict__ wb,
                const __bf16* __restrict__ xb,
                float* __restrict__ qkv) {
  int wid  = blockIdx.x * (blockDim.x >> 5) + (threadIdx.x >> 5);
  int lane = threadIdx.x & 31;
  int b    = wid / (48 * 16);
  int rem  = wid % (48 * 16);
  int o0   = (rem / 16) * 32;
  int n0   = (rem % 16) * 64;
  int m = lane & 15, h = lane >> 4;

  const __bf16* arow0 = wb + (size_t)(o0 + m) * C_;
  const __bf16* arow1 = wb + (size_t)(o0 + 16 + m) * C_;
  const __bf16* brow  = xb + ((size_t)b * N_ + n0 + m) * C_;  // +s*16*C_ folds

  f32x8 acc[2][4] = {};
  bf16x16 aA0, aA1, bA[4];     // fragment set A
  bf16x16 aB0, aB1, bB[4];     // fragment set B

  auto loadset = [&](bf16x16& a0f, bf16x16& a1f, bf16x16 (&bf)[4], int kk) {
    a0f = load_a_frag(arow0, kk, h);
    a1f = load_a_frag(arow1, kk, h);
#pragma unroll
    for (int s = 0; s < 4; ++s)
      bf[s] = load_b_frag(brow + (size_t)s * 16 * C_, kk, h);
  };
  auto domma = [&](bf16x16 a0f, bf16x16 a1f, bf16x16 (&bf)[4]) {
#pragma unroll
    for (int s = 0; s < 4; ++s) acc[0][s] = wmma_bf16(a0f, bf[s], acc[0][s]);
#pragma unroll
    for (int s = 0; s < 4; ++s) acc[1][s] = wmma_bf16(a1f, bf[s], acc[1][s]);
  };

  loadset(aA0, aA1, bA, 0);
  int k0 = 0;
  for (; k0 < C_ - 64; k0 += 64) {
    loadset(aB0, aB1, bB, k0 + 32);   // prefetch odd step
    domma(aA0, aA1, bA);              // consume even step
    loadset(aA0, aA1, bA, k0 + 64);   // prefetch next even step
    domma(aB0, aB1, bB);              // consume odd step
  }
  loadset(aB0, aB1, bB, k0 + 32);     // last odd step
  domma(aA0, aA1, bA);
  domma(aB0, aB1, bB);

#pragma unroll
  for (int t = 0; t < 2; ++t) {
#pragma unroll
    for (int s = 0; s < 4; ++s) {
      int n = n0 + s * 16 + m;           // D: col = lane&15, rows = r + 8h
      float* dst = qkv + ((size_t)(b * N_ + n)) * C3 + o0 + t * 16 + h * 8;
      *(float4*)(dst)     = make_float4(acc[t][s][0], acc[t][s][1],
                                        acc[t][s][2], acc[t][s][3]);
      *(float4*)(dst + 4) = make_float4(acc[t][s][4], acc[t][s][5],
                                        acc[t][s][6], acc[t][s][7]);
    }
  }
}

// ---------------- focus / relu / scale + operand repack ---------------------
__device__ __forceinline__ float block_sum(float v, float* red) {
#pragma unroll
  for (int off = 16; off > 0; off >>= 1) v += __shfl_xor(v, off, 32);
  int lane = threadIdx.x & 31, w = threadIdx.x >> 5;
  if (lane == 0) red[w] = v;
  __syncthreads();
  float r = (threadIdx.x < 8) ? red[threadIdx.x] : 0.0f;
  if (w == 0) {
#pragma unroll
    for (int off = 4; off > 0; off >>= 1) r += __shfl_xor(r, off, 32);
    if (lane == 0) red[0] = r;
  }
  __syncthreads();
  float out = red[0];
  __syncthreads();
  return out;
}

__global__ void focus_kernel(const float* __restrict__ qkv,
                             const float* __restrict__ scale_p,
                             const float* __restrict__ pos_enc,
                             __bf16* __restrict__ qb,   // (B*h, N, hd)
                             __bf16* __restrict__ kt,   // (B*h, hd, N)
                             __bf16* __restrict__ vt) { // (B*h, hd, N)
  __shared__ float red[8];
  int n = blockIdx.x & (N_ - 1);
  int b = blockIdx.x >> 10;
  int tid = threadIdx.x;
  const float* row = qkv + ((size_t)b * N_ + n) * C3;

  float tq[2], tk[2];
  float s2q = 0.f, s6q = 0.f, s2k = 0.f, s6k = 0.f;
#pragma unroll
  for (int e = 0; e < 2; ++e) {
    int c = tid + e * 256;
    float sp = scale_p[c];
    float sc = (sp > 20.f) ? sp : log1pf(__expf(sp));   // softplus
    float qv = row[c];
    float kv = row[C_ + c] + pos_enc[(size_t)n * C_ + c];
    qv = (fmaxf(qv, 0.f) + 1e-6f) / sc;
    kv = (fmaxf(kv, 0.f) + 1e-6f) / sc;
    tq[e] = qv; tk[e] = kv;
    float q3 = qv * qv * qv, k3 = kv * kv * kv;
    s2q += qv * qv; s6q += q3 * q3;
    s2k += kv * kv; s6k += k3 * k3;
  }
  s2q = block_sum(s2q, red);
  s6q = block_sum(s6q, red);
  s2k = block_sum(s2k, red);
  s6k = block_sum(s6k, red);
  float fq = sqrtf(s2q) * rsqrtf(s6q);   // ||t|| / ||t^3||
  float fk = sqrtf(s2k) * rsqrtf(s6k);

#pragma unroll
  for (int e = 0; e < 2; ++e) {
    int c = tid + e * 256;
    int hh = c >> 6, d = c & (HD - 1);
    int bh = b * HEADS + hh;
    float q3 = tq[e] * tq[e] * tq[e];
    float k3 = tk[e] * tk[e] * tk[e];
    qb[((size_t)bh * N_ + n) * HD + d]  = (__bf16)(q3 * fq);
    kt[((size_t)bh * HD + d) * N_ + n]  = (__bf16)(k3 * fk);
    vt[((size_t)bh * HD + d) * N_ + n]  = (__bf16)row[2 * C_ + c];
  }
}

// ---------------- fused depthwise 3x3 (lepe) + 5x5 (dwc) -------------------
__global__ void conv_kernel(const float* __restrict__ qkv,
                            const float* __restrict__ w_v,
                            const float* __restrict__ b_v,
                            const float* __restrict__ w_dwc,
                            const float* __restrict__ b_dwc,
                            float* __restrict__ convsum) { // (B,N,C)
  size_t t = (size_t)blockIdx.x * 256 + threadIdx.x;       // c fastest
  int c = (int)(t & (C_ - 1));
  int n = (int)((t >> 9) & (N_ - 1));
  int b = (int)(t >> 19);
  int y = n >> 5, x = n & 31;
  const float* vbase = qkv + (size_t)b * N_ * C3 + 2 * C_ + c;  // + nn*C3

  float lep = b_v[c];
#pragma unroll
  for (int dy = 0; dy < 3; ++dy) {
    int yy = y + dy - 1;
    if ((unsigned)yy < 32u) {
#pragma unroll
      for (int dx = 0; dx < 3; ++dx) {
        int xx = x + dx - 1;
        if ((unsigned)xx < 32u)
          lep += vbase[(size_t)(yy * 32 + xx) * C3] * w_v[c * 9 + dy * 3 + dx];
      }
    }
  }
  int cd = c & (HD - 1);
  float fm = b_dwc[cd];
#pragma unroll
  for (int dy = 0; dy < 5; ++dy) {
    int yy = y + dy - 2;
    if ((unsigned)yy < 32u) {
#pragma unroll
      for (int dx = 0; dx < 5; ++dx) {
        int xx = x + dx - 2;
        if ((unsigned)xx < 32u)
          fm += vbase[(size_t)(yy * 32 + xx) * C3] * w_dwc[cd * 25 + dy * 5 + dx];
      }
    }
  }
  convsum[t] = lep + fm;
}

// ---------------- ksum + z --------------------------------------------------
__global__ void ksum_kernel(const __bf16* __restrict__ kt,
                            float* __restrict__ ksum) {
  int t = blockIdx.x * 256 + threadIdx.x;        // (bh, c) : 256*64
  const __bf16* row = kt + (size_t)t * N_;
  float s = 0.f;
  for (int n = 0; n < N_; n += 8) {
    bf16x8 v = *(const bf16x8*)(row + n);
#pragma unroll
    for (int e = 0; e < 8; ++e) s += (float)v[e];
  }
  ksum[t] = s;
}

__global__ void z_kernel(const __bf16* __restrict__ qb,
                         const float* __restrict__ ksum,
                         float* __restrict__ z) {
  int t = blockIdx.x * 256 + threadIdx.x;        // (bh, i) : 256*1024
  int bh = t >> 10;
  const __bf16* qrow = qb + (size_t)t * HD;
  const float*  ks   = ksum + bh * HD;
  float s = 0.f;
#pragma unroll 8
  for (int c = 0; c < HD; ++c) s += (float)qrow[c] * ks[c];
  z[t] = 1.f / (s + 1e-6f);
}

// ---------------- kv = k^T v  (per bh: 64x64, K=1024) -----------------------
// wave tile: 32 (c) x 64 (d), ping-pong pipeline; 512 waves total
__global__ void __launch_bounds__(256, 1)
kv_gemm_kernel(const __bf16* __restrict__ kt,
               const __bf16* __restrict__ vt,
               __bf16* __restrict__ kvT) { // (bh, d, c)
  int wid  = blockIdx.x * (blockDim.x >> 5) + (threadIdx.x >> 5);
  int lane = threadIdx.x & 31;
  int bh = wid >> 1;
  int c0 = (wid & 1) * 32;
  int m = lane & 15, h = lane >> 4;

  const __bf16* arow0 = kt + ((size_t)bh * HD + c0 + m) * N_;
  const __bf16* arow1 = kt + ((size_t)bh * HD + c0 + 16 + m) * N_;
  const __bf16* brow  = vt + ((size_t)bh * HD + m) * N_;      // +s*16*N_ folds

  f32x8 acc[2][4] = {};
  bf16x16 aA0, aA1, bA[4];
  bf16x16 aB0, aB1, bB[4];

  auto loadset = [&](bf16x16& a0f, bf16x16& a1f, bf16x16 (&bf)[4], int kk) {
    a0f = load_a_frag(arow0, kk, h);
    a1f = load_a_frag(arow1, kk, h);
#pragma unroll
    for (int s = 0; s < 4; ++s)
      bf[s] = load_b_frag(brow + (size_t)s * 16 * N_, kk, h);
  };
  auto domma = [&](bf16x16 a0f, bf16x16 a1f, bf16x16 (&bf)[4]) {
#pragma unroll
    for (int s = 0; s < 4; ++s) acc[0][s] = wmma_bf16(a0f, bf[s], acc[0][s]);
#pragma unroll
    for (int s = 0; s < 4; ++s) acc[1][s] = wmma_bf16(a1f, bf[s], acc[1][s]);
  };

  loadset(aA0, aA1, bA, 0);
  int k0 = 0;
  for (; k0 < N_ - 64; k0 += 64) {
    loadset(aB0, aB1, bB, k0 + 32);
    domma(aA0, aA1, bA);
    loadset(aA0, aA1, bA, k0 + 64);
    domma(aB0, aB1, bB);
  }
  loadset(aB0, aB1, bB, k0 + 32);
  domma(aA0, aA1, bA);
  domma(aB0, aB1, bB);

#pragma unroll
  for (int t = 0; t < 2; ++t) {
#pragma unroll
    for (int s = 0; s < 4; ++s) {
      int d = s * 16 + m;
      __bf16* dst = kvT + ((size_t)bh * HD + d) * HD + c0 + t * 16 + h * 8;
#pragma unroll
      for (int r = 0; r < 8; ++r) dst[r] = (__bf16)acc[t][s][r];
    }
  }
}

// ---------------- out = (q @ kv) * z + convsum, written NCHW ----------------
// K=64: hoist all 10 fragment loads, then 8 back-to-back WMMAs
__global__ void __launch_bounds__(256, 1)
out_gemm_kernel(const __bf16* __restrict__ qb,
                const __bf16* __restrict__ kvT,
                const float* __restrict__ z,
                const float* __restrict__ convsum,
                float* __restrict__ out) { // (B, C, N)
  int wid  = blockIdx.x * (blockDim.x >> 5) + (threadIdx.x >> 5); // 16384 waves
  int lane = threadIdx.x & 31;
  int bh = wid >> 6;
  int i0 = (wid & 63) * 16;
  int b = bh >> 3, hh = bh & 7;
  int m = lane & 15, h = lane >> 4;

  const __bf16* arow = qb + ((size_t)bh * N_ + i0 + m) * HD;
  bf16x16 a0 = load_a_frag(arow, 0, h);
  bf16x16 a1 = load_a_frag(arow, 32, h);
  bf16x16 b0[4], b1[4];
#pragma unroll
  for (int s = 0; s < 4; ++s) {
    const __bf16* brow = kvT + ((size_t)bh * HD + s * 16 + m) * HD;
    b0[s] = load_b_frag(brow, 0, h);
    b1[s] = load_b_frag(brow, 32, h);
  }
  f32x8 acc[4] = {};
#pragma unroll
  for (int s = 0; s < 4; ++s) acc[s] = wmma_bf16(a0, b0[s], acc[s]);
#pragma unroll
  for (int s = 0; s < 4; ++s) acc[s] = wmma_bf16(a1, b1[s], acc[s]);

#pragma unroll
  for (int s = 0; s < 4; ++s) {
    int d  = s * 16 + m;
    int ch = hh * HD + d;
#pragma unroll
    for (int r = 0; r < 8; ++r) {
      int i = i0 + r + h * 8;
      float zv = z[bh * N_ + i];
      float cs = convsum[((size_t)(b * N_ + i)) * C_ + ch];
      out[((size_t)(b * C_ + ch)) * N_ + i] = acc[s][r] * zv + cs;
    }
  }
}

// ---------------------------------------------------------------------------
extern "C" void kernel_launch(void* const* d_in, const int* in_sizes, int n_in,
                              void* d_out, int out_size, void* d_ws, size_t ws_size,
                              hipStream_t stream) {
  const float* x       = (const float*)d_in[0];
  const float* w_qkv   = (const float*)d_in[1];
  const float* w_v     = (const float*)d_in[2];
  const float* b_v     = (const float*)d_in[3];
  const float* w_dwc   = (const float*)d_in[4];
  const float* b_dwc   = (const float*)d_in[5];
  const float* scale_p = (const float*)d_in[6];
  const float* pos_enc = (const float*)d_in[7];
  float* out = (float*)d_out;

  char* ws = (char*)d_ws;
  size_t off = 0;
  auto alloc = [&](size_t bytes) { char* p = ws + off; off += (bytes + 255) & ~(size_t)255; return p; };
  __bf16* wb      = (__bf16*)alloc((size_t)C3 * C_ * 2);           // 1.5 MB
  __bf16* xb      = (__bf16*)alloc((size_t)B_ * N_ * C_ * 2);      // 32 MB
  float*  qkv     = (float*) alloc((size_t)B_ * N_ * C3 * 4);      // 192 MB
  __bf16* qb      = (__bf16*)alloc((size_t)B_ * HEADS * N_ * HD * 2);  // 32 MB
  __bf16* kt      = (__bf16*)alloc((size_t)B_ * HEADS * HD * N_ * 2);  // 32 MB
  __bf16* vt      = (__bf16*)alloc((size_t)B_ * HEADS * HD * N_ * 2);  // 32 MB
  __bf16* kvT     = (__bf16*)alloc((size_t)B_ * HEADS * HD * HD * 2);  // 2 MB
  float*  ksum    = (float*) alloc((size_t)B_ * HEADS * HD * 4);
  float*  zbuf    = (float*) alloc((size_t)B_ * HEADS * N_ * 4);
  float*  convsum = (float*) alloc((size_t)B_ * N_ * C_ * 4);      // 64 MB

  // 1. operand conversion
  w_to_bf16_kernel<<<(C3 * C_ + 255) / 256, 256, 0, stream>>>(w_qkv, wb, C3 * C_);
  x_transpose_kernel<<<B_ * (C_ / 32) * (N_ / 32), 256, 0, stream>>>(x, xb);

  // 2. qkv GEMM: 48 o-supertiles * 16 n-supertiles * 32 batches waves, 8/block
  qkv_gemm_kernel<<<(48 * 16 * B_) / 8, 256, 0, stream>>>(wb, xb, qkv);

  // 3. relu/scale/focus + head repack (one block per (b,n))
  focus_kernel<<<B_ * N_, 256, 0, stream>>>(qkv, scale_p, pos_enc, qb, kt, vt);

  // 4. fused depthwise convs on v
  conv_kernel<<<(B_ * N_ * C_) / 256, 256, 0, stream>>>(qkv, w_v, b_v, w_dwc, b_dwc, convsum);

  // 5. normalizer
  ksum_kernel<<<(B_ * HEADS * HD) / 256, 256, 0, stream>>>(kt, ksum);
  z_kernel<<<(B_ * HEADS * N_) / 256, 256, 0, stream>>>(qb, ksum, zbuf);

  // 6. kv = k^T v  (256 bh * 2 c-supertiles waves)
  kv_gemm_kernel<<<(B_ * HEADS * 2) / 8, 256, 0, stream>>>(kt, vt, kvT);

  // 7. out = q@kv * z + convs, NCHW store
  out_gemm_kernel<<<(B_ * HEADS * 64) / 8, 256, 0, stream>>>(qb, kvT, zbuf, convsum, out);
}
